// DGCNN_40733469835823
// MI455X (gfx1250) — matrix operand order repaired and checked by reading, not compiled, not run on hardware
//
#include <hip/hip_runtime.h>
#include <hip/hip_bf16.h>

typedef __bf16 bf16;
typedef __attribute__((ext_vector_type(16))) __bf16 v16bf;
typedef __attribute__((ext_vector_type(8)))  float  v8f;

// ---------------------------------------------------------------------------
// WMMA helpers (CDNA5 wave32, V_WMMA_F32_16X16X32_BF16)
// ---------------------------------------------------------------------------
__device__ inline v8f zero8() {
    v8f z;
#pragma unroll
    for (int i = 0; i < 8; ++i) z[i] = 0.f;
    return z;
}

__device__ inline v8f wmma_bf16(v16bf a, v16bf b, v8f c) {
    // 8 args: (neg_a, A, neg_b, B, c_mod, C, reuse_a, reuse_b)
    return __builtin_amdgcn_wmma_f32_16x16x32_bf16(false, a, false, b, (short)0, c, false, false);
}

// A fragment: 16x32 bf16 tile from a row-major buffer (LDS or global).
// ISA layout: lane m=lane&15; lanes<16 hold K={0..7,16..23}, lanes>=16 hold K={8..15,24..31}.
__device__ inline v16bf load_a_frag(const bf16* __restrict__ base, int ld, int kc) {
    int lane = threadIdx.x & 31;
    int m = lane & 15;
    int kb = kc + ((lane & 16) ? 8 : 0);
    const bf16* r = base + m * ld + kb;
    v16bf a;
#pragma unroll
    for (int i = 0; i < 8; ++i) a[i] = r[i];
#pragma unroll
    for (int i = 0; i < 8; ++i) a[8 + i] = r[16 + i];
    return a;
}

// B fragment: 32x16 bf16 from a TRANSPOSED (N x K row-major) weight buffer.
// lane n=ncb+(lane&15); lanes<16 hold K={0..7,16..23}, lanes>=16 hold K={8..15,24..31}.
__device__ inline v16bf load_b_frag(const bf16* __restrict__ wt, int ldk, int ncb, int kc) {
    int lane = threadIdx.x & 31;
    int n = ncb + (lane & 15);
    int kb = kc + ((lane & 16) ? 8 : 0);
    const bf16* p = wt + (size_t)n * ldk + kb;
    v16bf b;
#pragma unroll
    for (int i = 0; i < 8; ++i) b[i] = p[i];
#pragma unroll
    for (int i = 0; i < 8; ++i) b[8 + i] = p[16 + i];
    return b;
}

// ---------------------------------------------------------------------------
// Weight conversion: f32 (R x C) -> bf16 transposed (C x KP), zero-padded K
// ---------------------------------------------------------------------------
__global__ void convert_wT_kernel(const float* __restrict__ src, bf16* __restrict__ dst,
                                  int R, int C, int KP, int total) {
    int t = blockIdx.x * 256 + threadIdx.x;
    if (t >= total) return;
    int n = t / KP, k = t - n * KP;
    float v = (k < R) ? src[k * C + n] : 0.f;
    dst[t] = (bf16)v;
}

// ---------------------------------------------------------------------------
// KNN: wave-per-point; distance row in LDS; 30 argmin passes; pad to 32 with self
// ---------------------------------------------------------------------------
template <int F>
__global__ __launch_bounds__(256) void knn_kernel(const float* __restrict__ xin,
                                                  int* __restrict__ idx) {
    __shared__ float s_d[8][1024];
    __shared__ float s_xi[8][F];
    int w = threadIdx.x >> 5, lane = threadIdx.x & 31;
    int pt = blockIdx.x * 8 + w;
    int g = pt >> 10, i = pt & 1023;
    const float* xg = xin + (size_t)g * 1024 * F;
    for (int k = lane; k < F; k += 32) s_xi[w][k] = xg[i * F + k];
    __syncthreads();
    for (int t = 0; t < 32; ++t) {
        int q = t * 32 + lane;
        const float* xq = xg + (size_t)q * F;
        float d = 0.f;
#pragma unroll
        for (int k = 0; k < F; ++k) { float df = s_xi[w][k] - xq[k]; d += df * df; }
        s_d[w][q] = d;
    }
    int* op = idx + (size_t)pt * 32;
    for (int s = 0; s < 30; ++s) {
        float best = 3.0e38f; int bq = 1 << 30;
        for (int t = 0; t < 32; ++t) {
            int q = t * 32 + lane;
            float v = s_d[w][q];
            if (v < best || (v == best && q < bq)) { best = v; bq = q; }
        }
#pragma unroll
        for (int o = 16; o > 0; o >>= 1) {
            float ov = __shfl_xor(best, o, 32);
            int   oq = __shfl_xor(bq, o, 32);
            if (ov < best || (ov == best && oq < bq)) { best = ov; bq = oq; }
        }
        if (lane == 0) { op[s] = bq; s_d[w][bq] = 3.0e38f; }
    }
    if (lane == 0) { op[30] = i; op[31] = i; }  // self-pad: unchanged max-pool
}

// ---------------------------------------------------------------------------
// EdgeConv: 4 points/block (128 thr = 4 waves); wave owns ALL 32 padded
// neighbors of one point = two 16-row M-tiles sharing every B-fragment
// (2x weight reuse per L2 byte), max-pool finishes fully in-wave.
// e=[xi, xj-xi] -> relu(e@W1+b1) -> relu(@W2+b2) -> max over 32 neighbors
// ---------------------------------------------------------------------------
template <int F>
__global__ __launch_bounds__(128) void edgeconv_kernel(
    const float* __restrict__ xin, const int* __restrict__ idx,
    const bf16* __restrict__ w1t, const float* __restrict__ b1,
    const bf16* __restrict__ w2t, const float* __restrict__ b2,
    float* __restrict__ xout) {
    constexpr int KE = (2 * F + 31) & ~31;   // 32 (F=14) or 128 (F=64)
    constexpr int KC1 = KE / 32;
    __shared__ bf16 s_e[4][32 * KE];
    __shared__ bf16 s_h1[4][32 * 64];
    __shared__ int  s_j[4][32];
    int w = threadIdx.x >> 5, lane = threadIdx.x & 31;
    int p = blockIdx.x * 4 + w;
    int g = p >> 10, ig = p & 1023;
    const float* xg = xin + (size_t)g * 1024 * F;
    s_j[w][lane] = idx[(size_t)p * 32 + lane];
    __syncthreads();
    for (int e = lane; e < 32 * KE; e += 32) {
        int r = e / KE, k = e % KE;
        float v;
        if (k < F) v = xg[ig * F + k];
        else if (k < 2 * F) {
            int kk = k - F;
            v = xg[(size_t)s_j[w][r] * F + kk] - xg[ig * F + kk];
        } else v = 0.f;
        s_e[w][e] = (bf16)v;
    }
    __syncthreads();
    // ---- MLP1: (32 x KE) @ (KE x 64), two M-tiles per B-frag ----
    v16bf a0[KC1], a1[KC1];
#pragma unroll
    for (int kc = 0; kc < KC1; ++kc) {
        a0[kc] = load_a_frag(s_e[w], KE, kc * 32);
        a1[kc] = load_a_frag(s_e[w] + 16 * KE, KE, kc * 32);
    }
    v8f acc0[4], acc1[4];
#pragma unroll
    for (int n = 0; n < 4; ++n) { acc0[n] = zero8(); acc1[n] = zero8(); }
#pragma unroll
    for (int kc = 0; kc < KC1; ++kc)
#pragma unroll
        for (int n = 0; n < 4; ++n) {
            v16bf b = load_b_frag(w1t, KE, n * 16, kc * 32);
            acc0[n] = wmma_bf16(a0[kc], b, acc0[n]);
            acc1[n] = wmma_bf16(a1[kc], b, acc1[n]);
        }
#pragma unroll
    for (int n = 0; n < 4; ++n) {
        int nc = n * 16 + (lane & 15);
        float bias = b1[nc];
#pragma unroll
        for (int v = 0; v < 8; ++v) {
            int m = v + ((lane & 16) ? 8 : 0);
            s_h1[w][m * 64 + nc]        = (bf16)fmaxf(acc0[n][v] + bias, 0.f);
            s_h1[w][(m + 16) * 64 + nc] = (bf16)fmaxf(acc1[n][v] + bias, 0.f);
        }
    }
    __syncthreads();
    // ---- MLP2: (32 x 64) @ (64 x 64), two M-tiles per B-frag ----
    v16bf c0[2], c1[2];
    c0[0] = load_a_frag(s_h1[w], 64, 0);
    c0[1] = load_a_frag(s_h1[w], 64, 32);
    c1[0] = load_a_frag(s_h1[w] + 16 * 64, 64, 0);
    c1[1] = load_a_frag(s_h1[w] + 16 * 64, 64, 32);
#pragma unroll
    for (int n = 0; n < 4; ++n) { acc0[n] = zero8(); acc1[n] = zero8(); }
#pragma unroll
    for (int kc = 0; kc < 2; ++kc)
#pragma unroll
        for (int n = 0; n < 4; ++n) {
            v16bf b = load_b_frag(w2t, 64, n * 16, kc * 32);
            acc0[n] = wmma_bf16(c0[kc], b, acc0[n]);
            acc1[n] = wmma_bf16(c1[kc], b, acc1[n]);
        }
    // bias + relu + max over all 32 neighbor rows, fully in-wave
#pragma unroll
    for (int n = 0; n < 4; ++n) {
        int nc = n * 16 + (lane & 15);
        float bias = b2[nc];
        float mx = -3.0e38f;
#pragma unroll
        for (int v = 0; v < 8; ++v) {
            mx = fmaxf(mx, fmaxf(acc0[n][v] + bias, 0.f));
            mx = fmaxf(mx, fmaxf(acc1[n][v] + bias, 0.f));
        }
        mx = fmaxf(mx, __shfl_xor(mx, 16, 32));
        if (lane < 16) xout[(size_t)p * 64 + nc] = mx;
    }
}

// ---------------------------------------------------------------------------
// Fused head MLP: 16-row tile per block (128 thr = 4 waves); h0 kept in LDS bf16
// 192 -> 1024 -> 256 -> 128 -> 64 (relu on all but last)
// ---------------------------------------------------------------------------
__global__ __launch_bounds__(128) void head_kernel(
    const float* __restrict__ x1, const float* __restrict__ x2, const float* __restrict__ x3,
    const bf16* __restrict__ lwT, const float* __restrict__ lb,
    const bf16* __restrict__ w1T, const float* __restrict__ hb1,
    const bf16* __restrict__ w2T, const float* __restrict__ hb2,
    const bf16* __restrict__ w3T, const float* __restrict__ hb3,
    float* __restrict__ out) {
    __shared__ bf16 s_in[16 * 192];
    __shared__ bf16 s_h0[16 * 1024];
    __shared__ bf16 s_h1[16 * 256];
    __shared__ bf16 s_h2[16 * 128];
    int tid = threadIdx.x, w = tid >> 5, lane = tid & 31;
    int row0 = blockIdx.x * 16;
    for (int e = tid; e < 16 * 192; e += 128) {
        int m = e / 192, k = e - m * 192;
        size_t r = (size_t)(row0 + m) * 64;
        float v = (k < 64) ? x1[r + k] : (k < 128) ? x2[r + k - 64] : x3[r + k - 128];
        s_in[e] = (bf16)v;
    }
    __syncthreads();
    // stage 1: 192 -> 1024 (wave handles 256 output cols)
    {
        v16bf a[6];
#pragma unroll
        for (int kc = 0; kc < 6; ++kc) a[kc] = load_a_frag(s_in, 192, kc * 32);
        for (int nt = 0; nt < 16; ++nt) {
            int ncb = (w * 16 + nt) * 16;
            v8f acc = zero8();
#pragma unroll
            for (int kc = 0; kc < 6; ++kc)
                acc = wmma_bf16(a[kc], load_b_frag(lwT, 192, ncb, kc * 32), acc);
            int nc = ncb + (lane & 15);
            float bias = lb[nc];
#pragma unroll
            for (int v = 0; v < 8; ++v) {
                int m = v + ((lane & 16) ? 8 : 0);
                s_h0[m * 1024 + nc] = (bf16)fmaxf(acc[v] + bias, 0.f);
            }
        }
    }
    __syncthreads();
    // stage 2: 1024 -> 256
    {
        v8f acc[4];
#pragma unroll
        for (int n = 0; n < 4; ++n) acc[n] = zero8();
        for (int kc = 0; kc < 32; ++kc) {
            v16bf a = load_a_frag(s_h0, 1024, kc * 32);
#pragma unroll
            for (int n = 0; n < 4; ++n)
                acc[n] = wmma_bf16(a, load_b_frag(w1T, 1024, (w * 4 + n) * 16, kc * 32), acc[n]);
        }
#pragma unroll
        for (int n = 0; n < 4; ++n) {
            int nc = (w * 4 + n) * 16 + (lane & 15);
            float bias = hb1[nc];
#pragma unroll
            for (int v = 0; v < 8; ++v) {
                int m = v + ((lane & 16) ? 8 : 0);
                s_h1[m * 256 + nc] = (bf16)fmaxf(acc[n][v] + bias, 0.f);
            }
        }
    }
    __syncthreads();
    // stage 3: 256 -> 128
    {
        v8f acc[2];
        acc[0] = zero8(); acc[1] = zero8();
        for (int kc = 0; kc < 8; ++kc) {
            v16bf a = load_a_frag(s_h1, 256, kc * 32);
#pragma unroll
            for (int n = 0; n < 2; ++n)
                acc[n] = wmma_bf16(a, load_b_frag(w2T, 256, (w * 2 + n) * 16, kc * 32), acc[n]);
        }
#pragma unroll
        for (int n = 0; n < 2; ++n) {
            int nc = (w * 2 + n) * 16 + (lane & 15);
            float bias = hb2[nc];
#pragma unroll
            for (int v = 0; v < 8; ++v) {
                int m = v + ((lane & 16) ? 8 : 0);
                s_h2[m * 128 + nc] = (bf16)fmaxf(acc[n][v] + bias, 0.f);
            }
        }
    }
    __syncthreads();
    // stage 4: 128 -> 64 (no relu)
    {
        v8f acc = zero8();
#pragma unroll
        for (int kc = 0; kc < 4; ++kc)
            acc = wmma_bf16(load_a_frag(s_h2, 128, kc * 32),
                            load_b_frag(w3T, 128, w * 16, kc * 32), acc);
        int nc = w * 16 + (lane & 15);
        float bias = hb3[nc];
#pragma unroll
        for (int v = 0; v < 8; ++v) {
            int m = v + ((lane & 16) ? 8 : 0);
            out[(size_t)(row0 + m) * 64 + nc] = acc[v] + bias;
        }
    }
}

// ---------------------------------------------------------------------------
extern "C" void kernel_launch(void* const* d_in, const int* in_sizes, int n_in,
                              void* d_out, int out_size, void* d_ws, size_t ws_size,
                              hipStream_t stream) {
    (void)in_sizes; (void)n_in; (void)out_size; (void)ws_size;
    const float* x    = (const float*)d_in[0];
    const float* c1w1 = (const float*)d_in[2];
    const float* c1b1 = (const float*)d_in[3];
    const float* c1w2 = (const float*)d_in[4];
    const float* c1b2 = (const float*)d_in[5];
    const float* c2w1 = (const float*)d_in[6];
    const float* c2b1 = (const float*)d_in[7];
    const float* c2w2 = (const float*)d_in[8];
    const float* c2b2 = (const float*)d_in[9];
    const float* c3w1 = (const float*)d_in[10];
    const float* c3b1 = (const float*)d_in[11];
    const float* c3w2 = (const float*)d_in[12];
    const float* c3b2 = (const float*)d_in[13];
    const float* lw   = (const float*)d_in[14];
    const float* lb   = (const float*)d_in[15];
    const float* hw1  = (const float*)d_in[16];
    const float* hb1  = (const float*)d_in[17];
    const float* hw2  = (const float*)d_in[18];
    const float* hb2  = (const float*)d_in[19];
    const float* hw3  = (const float*)d_in[20];
    const float* hb3  = (const float*)d_in[21];
    float* out = (float*)d_out;

    char* ws = (char*)d_ws;
    size_t off = 0;
    auto alloc = [&](size_t bytes) {
        char* p = ws + off;
        off += (bytes + 255) & ~(size_t)255;
        return p;
    };
    int*   idxb  = (int*)  alloc(32768ull * 32 * 4);
    float* x1    = (float*)alloc(32768ull * 64 * 4);
    float* x2    = (float*)alloc(32768ull * 64 * 4);
    float* x3    = (float*)alloc(32768ull * 64 * 4);
    bf16* c1w1T = (bf16*)alloc(64ull * 32 * 2);
    bf16* c1w2T = (bf16*)alloc(64ull * 64 * 2);
    bf16* c2w1T = (bf16*)alloc(64ull * 128 * 2);
    bf16* c2w2T = (bf16*)alloc(64ull * 64 * 2);
    bf16* c3w1T = (bf16*)alloc(64ull * 128 * 2);
    bf16* c3w2T = (bf16*)alloc(64ull * 64 * 2);
    bf16* lwT   = (bf16*)alloc(1024ull * 192 * 2);
    bf16* h1T   = (bf16*)alloc(256ull * 1024 * 2);
    bf16* h2T   = (bf16*)alloc(128ull * 256 * 2);
    bf16* h3T   = (bf16*)alloc(64ull * 128 * 2);

    auto conv = [&](const float* src, bf16* dst, int R, int C, int KP) {
        int total = C * KP;
        convert_wT_kernel<<<(total + 255) / 256, 256, 0, stream>>>(src, dst, R, C, KP, total);
    };
    conv(c1w1, c1w1T, 28, 64, 32);
    conv(c1w2, c1w2T, 64, 64, 64);
    conv(c2w1, c2w1T, 128, 64, 128);
    conv(c2w2, c2w2T, 64, 64, 64);
    conv(c3w1, c3w1T, 128, 64, 128);
    conv(c3w2, c3w2T, 64, 64, 64);
    conv(lw,   lwT, 192, 1024, 192);
    conv(hw1,  h1T, 1024, 256, 1024);
    conv(hw2,  h2T, 256, 128, 256);
    conv(hw3,  h3T, 128, 64, 128);

    knn_kernel<14><<<4096, 256, 0, stream>>>(x, idxb);
    edgeconv_kernel<14><<<8192, 128, 0, stream>>>(x, idxb, c1w1T, c1b1, c1w2T, c1b2, x1);
    knn_kernel<64><<<4096, 256, 0, stream>>>(x1, idxb);
    edgeconv_kernel<64><<<8192, 128, 0, stream>>>(x1, idxb, c2w1T, c2b1, c2w2T, c2b2, x2);
    knn_kernel<64><<<4096, 256, 0, stream>>>(x2, idxb);
    edgeconv_kernel<64><<<8192, 128, 0, stream>>>(x2, idxb, c3w1T, c3b1, c3w2T, c3b2, x3);
    head_kernel<<<2048, 128, 0, stream>>>(x1, x2, x3, lwT, lb, h1T, hb1, h2T, hb2, h3T, hb3, out);
}
